// MoSRAHRouter_49941879718135
// MI455X (gfx1250) — matrix operand order repaired and checked by reading, not compiled, not run on hardware
//
#include <hip/hip_runtime.h>
#include <hip/hip_bf16.h>

// ---------------------------------------------------------------------------
// MoE router for MI455X (gfx1250, wave32, WMMA).
//   logits = x @ W_r^T   (16384 x 2048) x (2048 x 64)
//   softmax / biased-softmax top-8, renormalized probs, expert histogram,
//   load-balance loss + max violation.
// GEMM is done with v_wmma_f32_16x16x32_bf16 (f32 accumulate); x is streamed
// once from HBM (134 MB dominates; problem is bandwidth bound), W_r is staged
// per-K-slab into LDS as bf16 and reused by all 8 waves of the block.
// ---------------------------------------------------------------------------

typedef __attribute__((ext_vector_type(16))) __bf16 v16bf;
typedef __attribute__((ext_vector_type(8)))  float  v8f;

#define H_DIM 2048
#define L_EXP 64
#define K_SEL 8
#define ROWS_PER_BLOCK 128          // 8 waves x 16 rows
#define KSLAB 512                   // K-slab of W_r staged in LDS per pass
#define WPAD 520                    // bf16 elements per W row in LDS (pad: no bank conflicts)
#define LPAD 68                     // floats per logits row in LDS (pad)
#define NTOK 16384                  // B*N
#define SEL_OFS 0                   // out offsets (floats)
#define PROB_OFS 131072
#define SCALAR_OFS 262144

// LDS layout (dynamic):
//   Wlds    : 64 * 520 bf16  = 66560 B
//   logitsS : 128 * 68 f32   = 34816 B
//   cntS    : 64 f32         =   256 B
//   biasS   : 64 f32         =   256 B
#define LDS_BYTES (66560 + 34816 + 256 + 256)

__global__ void router_zero(float* g) {
    if (threadIdx.x < L_EXP) g[threadIdx.x] = 0.0f;
}

__global__ void __launch_bounds__(256)
router_main(const float* __restrict__ x,
            const float* __restrict__ Wr,
            const float* __restrict__ bias,
            const unsigned char* __restrict__ mask,
            float* __restrict__ out,
            float* __restrict__ gcnts)
{
    extern __shared__ char smem[];
    __bf16* Wlds    = (__bf16*)smem;
    float*  logitsS = (float*)(smem + 66560);
    float*  cntS    = (float*)(smem + 66560 + 34816);
    float*  biasS   = cntS + L_EXP;

    const int tid  = threadIdx.x;
    const int wv   = tid >> 5;          // wave id 0..7
    const int lane = tid & 31;
    const int l16  = lane & 15;
    const bool hi  = (lane & 16) != 0;  // upper half-wave

    if (tid < L_EXP) { cntS[tid] = 0.0f; biasS[tid] = bias[tid]; }

    const int rowA = blockIdx.x * ROWS_PER_BLOCK + wv * 16 + l16;
    const float* aRow = x + (size_t)rowA * H_DIM;

    v8f c[4] = {};  // four 16x16 f32 accumulator tiles (experts 0-15,16-31,32-47,48-63)

    for (int s = 0; s < H_DIM / KSLAB; ++s) {
        __syncthreads();  // protect Wlds (and cntS/biasS init on first pass)

        // ---- cooperative stage of W_r K-slab into LDS, fp32 -> bf16 ----
        const float* wsrc = Wr + s * KSLAB;
        #pragma unroll
        for (int i = 0; i < (L_EXP * KSLAB / 2) / 256; ++i) {   // 64 float2 per thread
            int p  = tid + i * 256;
            int n  = p >> 8;             // 256 float2 pairs per expert row
            int kk = (p & 255) * 2;
            float2 f = *(const float2*)(wsrc + (size_t)n * H_DIM + kk);
            union { __bf16 h[2]; unsigned u; } pk;
            pk.h[0] = (__bf16)f.x; pk.h[1] = (__bf16)f.y;
            *(unsigned*)&Wlds[n * WPAD + kk] = pk.u;
        }
        __syncthreads();

        // ---- WMMA over this K-slab ----
        for (int kc = 0; kc < KSLAB / 32; ++kc) {
            // A fragment (16x32 bf16): lanes 0-15 row=l16, K {0..7,16..23};
            // lanes 16-31 row=l16, K {8..15,24..31}  (CDNA5 16-bit A layout)
            const int kOff = s * KSLAB + kc * 32 + (hi ? 8 : 0);
            const float* ap = aRow + kOff;
            __builtin_prefetch(ap + 64, 0, 1);   // global_prefetch for upcoming x
            float4 f0 = *(const float4*)(ap);
            float4 f1 = *(const float4*)(ap + 4);
            float4 f2 = *(const float4*)(ap + 16);
            float4 f3 = *(const float4*)(ap + 20);
            v16bf a;
            a[0]=(__bf16)f0.x;  a[1]=(__bf16)f0.y;  a[2]=(__bf16)f0.z;  a[3]=(__bf16)f0.w;
            a[4]=(__bf16)f1.x;  a[5]=(__bf16)f1.y;  a[6]=(__bf16)f1.z;  a[7]=(__bf16)f1.w;
            a[8]=(__bf16)f2.x;  a[9]=(__bf16)f2.y;  a[10]=(__bf16)f2.z; a[11]=(__bf16)f2.w;
            a[12]=(__bf16)f3.x; a[13]=(__bf16)f3.y; a[14]=(__bf16)f3.z; a[15]=(__bf16)f3.w;

            // B fragments (32x16 bf16): lane = expert column; lanes 0-15 K {0..7,16..23},
            // lanes 16-31 K {8..15,24..31} (matches A layout)
            const int klocal = kc * 32 + (hi ? 8 : 0);
            #pragma unroll
            for (int t = 0; t < 4; ++t) {
                const __bf16* bp = Wlds + (size_t)(t * 16 + l16) * WPAD + klocal;
                union { uint4 q[2]; v16bf v; } b;
                b.q[0] = *(const uint4*)(bp);
                b.q[1] = *(const uint4*)(bp + 16);
                c[t] = __builtin_amdgcn_wmma_f32_16x16x32_bf16(
                          /*neg_a=*/false, a, /*neg_b=*/false, b.v,
                          /*c_mod=*/(short)0, c[t],
                          /*reuse_a=*/false, /*reuse_b=*/false);
            }
        }
    }

    // ---- spill 16x64 logit tile to LDS (C/D layout: vgpr v = row, lane = col) ----
    const int rbase = hi ? 8 : 0;
    #pragma unroll
    for (int t = 0; t < 4; ++t)
        #pragma unroll
        for (int v = 0; v < 8; ++v)
            logitsS[(size_t)(wv * 16 + rbase + v) * LPAD + t * 16 + l16] = c[t][v];
    __syncthreads();

    // ---- per-token softmax + biased top-8: lanes 0..15 each own one row ----
    if (lane < 16) {
        const int g = blockIdx.x * ROWS_PER_BLOCK + wv * 16 + lane;
        float z[L_EXP];
        #pragma unroll
        for (int j = 0; j < L_EXP; ++j)
            z[j] = logitsS[(size_t)(wv * 16 + lane) * LPAD + j];

        float m1 = z[0];
        #pragma unroll
        for (int j = 1; j < L_EXP; ++j) m1 = fmaxf(m1, z[j]);

        // biased logits for selection (in place)
        #pragma unroll
        for (int j = 0; j < L_EXP; ++j) z[j] += biasS[j];

        unsigned long long taken = 0ull;
        float pk[K_SEL]; int idx[K_SEL]; float esum = 0.0f;
        #pragma unroll
        for (int k = 0; k < K_SEL; ++k) {
            float bv = -3.4e38f; int bi = 0;
            #pragma unroll
            for (int j = 0; j < L_EXP; ++j) {
                bool t = (taken >> j) & 1ull;
                if (!t && z[j] > bv) { bv = z[j]; bi = j; }
            }
            taken |= (1ull << bi);
            float zu = bv - biasS[bi];          // back to unbiased logit
            float e  = __expf(zu - m1);         // softmax denom cancels under renorm
            pk[k] = e; esum += e; idx[k] = bi;
        }

        const float am  = mask[g] ? 1.0f : 0.0f;
        const float inv = 1.0f / esum;
        #pragma unroll
        for (int k = 0; k < K_SEL; ++k) {
            out[SEL_OFS  + (size_t)g * K_SEL + k] = (float)idx[k];
            out[PROB_OFS + (size_t)g * K_SEL + k] = pk[k] * inv;
            atomicAdd(&cntS[idx[k]], am);
        }
    }
    __syncthreads();
    if (tid < L_EXP) atomicAdd(&gcnts[tid], cntS[tid]);
}

__global__ void router_reduce(const unsigned char* __restrict__ mask,
                              const float* __restrict__ gcnts,
                              float* __restrict__ out)
{
    __shared__ float red[256];
    __shared__ float dv[L_EXP];
    const int tid = threadIdx.x;

    float s = 0.0f;
    for (int i = tid; i < NTOK; i += 256) s += mask[i] ? 1.0f : 0.0f;
    red[tid] = s;
    __syncthreads();
    for (int off = 128; off > 0; off >>= 1) {
        if (tid < off) red[tid] += red[tid + off];
        __syncthreads();
    }
    const float denom = red[0] * (float)K_SEL;   // num_active_assignments
    if (tid < L_EXP) dv[tid] = gcnts[tid] / denom - (1.0f / (float)L_EXP);
    __syncthreads();
    if (tid == 0) {
        float sq = 0.0f, mx = -3.4e38f;
        for (int l = 0; l < L_EXP; ++l) { float d = dv[l]; sq += d * d; mx = fmaxf(mx, d); }
        out[SCALAR_OFS + 0] = (float)L_EXP * sq;   // load_balance_loss
        out[SCALAR_OFS + 1] = (float)L_EXP * mx;   // max_vio
    }
}

extern "C" void kernel_launch(void* const* d_in, const int* in_sizes, int n_in,
                              void* d_out, int out_size, void* d_ws, size_t ws_size,
                              hipStream_t stream) {
    const float*         x    = (const float*)d_in[0];         // (4,4096,2048) f32
    const float*         Wr   = (const float*)d_in[1];         // (64,2048) f32
    const float*         bias = (const float*)d_in[2];         // (64,) f32
    const unsigned char* mask = (const unsigned char*)d_in[3]; // (4,4096) bool
    float* out   = (float*)d_out;                              // 262146 f32
    float* gcnts = (float*)d_ws;                               // 64 f32 histogram

    (void)in_sizes; (void)n_in; (void)out_size; (void)ws_size;
    (void)hipFuncSetAttribute((const void*)router_main,
                              hipFuncAttributeMaxDynamicSharedMemorySize, LDS_BYTES);

    router_zero<<<1, 64, 0, stream>>>(gcnts);
    router_main<<<dim3(NTOK / ROWS_PER_BLOCK), dim3(256), LDS_BYTES, stream>>>(
        x, Wr, bias, mask, out, gcnts);
    router_reduce<<<1, 256, 0, stream>>>(mask, gcnts, out);
}
